// LSTMOutputProj_40793599377846
// MI455X (gfx1250) — matrix operand order repaired and checked by reading, not compiled, not run on hardware
//
#include <hip/hip_runtime.h>
#include <hip/hip_bf16.h>

typedef __bf16 bf16_t;
typedef __attribute__((ext_vector_type(16))) __bf16 v16bf;
typedef __attribute__((ext_vector_type(8)))  __bf16 v8bf;
typedef __attribute__((ext_vector_type(8)))  float  v8f;
typedef __attribute__((ext_vector_type(4)))  unsigned int v4u;
typedef __attribute__((ext_vector_type(8)))  int v8i_;
typedef __attribute__((ext_vector_type(4)))  int v4i_;

#define BT    512
#define EDIM  512
#define HDIM  1024
#define H4    4096
#define VB    32
#define NOUT  33
#define NSTEP 128

__device__ __forceinline__ bf16_t f32_to_bf16(float f) {
  unsigned u = __float_as_uint(f);
  u += 0x7FFFu + ((u >> 16) & 1u);
  unsigned short s = (unsigned short)(u >> 16);
  return __builtin_bit_cast(bf16_t, s);
}

__device__ __forceinline__ v16bf combine16(v8bf lo, v8bf hi) {
  v16bf r;
#pragma unroll
  for (int i = 0; i < 8; ++i) { r[i] = lo[i]; r[i + 8] = hi[i]; }
  return r;
}

__device__ __forceinline__ float sigmoidf_(float x) {
  return 1.0f / (1.0f + __expf(-x));
}

// ---- TDM: DMA one 128-row x 32-col bf16 tile (row stride K elems) into LDS ----
// LDS layout gets a 4-DWORD pad every 16 DWORDs -> 40-element row stride,
// matching the As/Bs [128][40] anti-bank-conflict layout.
__device__ __forceinline__ void tdm_load_tile(const bf16_t* gsrc, unsigned ldsOff, int K)
{
  unsigned long long ga = (unsigned long long)(uintptr_t)gsrc;
  v4u g0;
  g0[0] = 1u;                                            // count=1, user mode
  g0[1] = ldsOff;                                        // lds_addr (bytes)
  g0[2] = (unsigned)(ga & 0xFFFFFFFFu);                  // global_addr[31:0]
  g0[3] = (unsigned)((ga >> 32) & 0x01FFFFFFu)           // global_addr[56:32]
        | (2u << 30);                                    // type=2 ("image")
  v8i_ g1;
  g1[0] = (int)((1u << 16)       // data_size = 2 bytes
              | (1u << 20)       // pad_enable
              | (3u << 22)       // pad_interval: 16 DWORDs
              | (3u << 25));     // pad_amount:   4 DWORDs
  g1[1] = (int)(((unsigned)K & 0xFFFFu) << 16);          // tensor_dim0[15:0] @ bits[63:48]
  g1[2] = (int)((((unsigned)K >> 16) & 0xFFFFu)          // tensor_dim0[31:16]
              | (0u << 16));                             // tensor_dim1[15:0] (dim1 = 1<<20)
  g1[3] = (int)(16u | (32u << 16));                      // tensor_dim1[31:16]=16, tile_dim0=32
  g1[4] = (int)(128u);                                   // tile_dim1=128, tile_dim2=0
  g1[5] = (int)(unsigned)K;                              // tensor_dim0_stride[31:0]
  g1[6] = 0;                                             // stride[47:32]=0, dim1_stride lo
  g1[7] = 0;
  v4i_ gz  = {0, 0, 0, 0};
  v8i_ gz8 = {0, 0, 0, 0, 0, 0, 0, 0};
  __builtin_amdgcn_tensor_load_to_lds(g0, g1, gz, gz, gz8, 0);
}

// -------------------------------------------------------------------------
// Tiled bf16 WMMA GEMM: C[M,N] = A[M,K] * B[K,N] (+ bias[N]) (+ base[M,N])
//                              (+ cur[M,2] @ Weff[2,N])
// A row-major bf16, Bt = B^T row-major bf16 ([N,K]).
// Optional second (A1,Bt1) pair accumulates into the same C tile.
// 256 threads (8 waves), tile 128x128, wave tile 64x32 (4x2 wmma).
// Tiles staged by the Tensor Data Mover, double-buffered in LDS.
// -------------------------------------------------------------------------
__global__ __launch_bounds__(256)
void gemm_bf16_wmma(const bf16_t* __restrict__ A0, const bf16_t* __restrict__ Bt0,
                    const bf16_t* __restrict__ A1, const bf16_t* __restrict__ Bt1,
                    const float* __restrict__ bias, const float* __restrict__ base,
                    const float* __restrict__ cur2, const float* __restrict__ Weff,
                    float* __restrict__ C, int M, int N, int K)
{
  __shared__ __align__(16) bf16_t As[2][128][40];   // 40 = 32 data + 8 pad (TDM pads)
  __shared__ __align__(16) bf16_t Bs[2][128][40];

  const int tid  = threadIdx.x;
  const int lane = tid & 31;
  const int wave = tid >> 5;
  const int wm   = wave >> 2;        // 0..1 : 64 rows
  const int wn   = wave & 3;         // 0..3 : 32 cols
  const int mBase = blockIdx.y * 128;
  const int nBase = blockIdx.x * 128;

  v8f acc[4][2];
#pragma unroll
  for (int mi = 0; mi < 4; ++mi)
#pragma unroll
    for (int ni = 0; ni < 2; ++ni)
#pragma unroll
      for (int v = 0; v < 8; ++v) acc[mi][ni][v] = 0.0f;

  const int nck   = K >> 5;                    // chunks per pair
  const int total = nck * (A1 ? 2 : 1);

  // issue both tile DMAs for chunk c into LDS buffer `bsel` (wave 0 only)
  auto issue = [&](int c, int bsel) {
    const bf16_t* A  = (c < nck) ? A0 : A1;
    const bf16_t* Bt = (c < nck) ? Bt0 : Bt1;
    int kc = ((c < nck) ? c : c - nck) << 5;
    tdm_load_tile(A  + (size_t)mBase * K + kc,
                  (unsigned)(uintptr_t)&As[bsel][0][0], K);
    tdm_load_tile(Bt + (size_t)nBase * K + kc,
                  (unsigned)(uintptr_t)&Bs[bsel][0][0], K);
  };

  if (wave == 0) issue(0, 0);

  int buf = 0;
#pragma unroll 1
  for (int c = 0; c < total; ++c) {
    if (wave == 0) __builtin_amdgcn_s_wait_tensorcnt(0);   // chunk c resident
    __syncthreads();                                       // visible to all waves;
                                                           // also: prior reads of buf^1 done
    if (wave == 0 && c + 1 < total) issue(c + 1, buf ^ 1); // DMA next chunk under compute

    // A fragments: 16x32 bf16; lanes 0-15 hold K {0..7,16..23}, 16-31 hold {8..15,24..31}
    v16bf af[4];
#pragma unroll
    for (int mi = 0; mi < 4; ++mi) {
      int r   = wm * 64 + mi * 16 + (lane & 15);
      int klo = (lane >> 4) << 3;
      v8bf lo = *(const v8bf*)&As[buf][r][klo];
      v8bf hi = *(const v8bf*)&As[buf][r][16 + klo];
      af[mi] = combine16(lo, hi);
    }
    // B fragments: 32x16; lanes 0-15 col=lane K {0..7,16..23}; lanes 16-31 col=lane-16 K {8..15,24..31}
    v16bf bfr[2];
#pragma unroll
    for (int ni = 0; ni < 2; ++ni) {
      int r  = wn * 32 + ni * 16 + (lane & 15);
      int kb = (lane >> 4) << 3;
      v8bf lo = *(const v8bf*)&Bs[buf][r][kb];
      v8bf hi = *(const v8bf*)&Bs[buf][r][kb + 16];
      bfr[ni] = combine16(lo, hi);
    }
#pragma unroll
    for (int mi = 0; mi < 4; ++mi)
#pragma unroll
      for (int ni = 0; ni < 2; ++ni)
        acc[mi][ni] = __builtin_amdgcn_wmma_f32_16x16x32_bf16(
            false, af[mi], false, bfr[ni], (short)0, acc[mi][ni], false, false);

    __syncthreads();           // all waves done reading `buf` before it is re-DMA'd
    buf ^= 1;
  }

  // epilogue: C 16x16 layout -> VGPR v: M = v + (lane>=16)*8, N = lane&15
#pragma unroll
  for (int mi = 0; mi < 4; ++mi) {
#pragma unroll
    for (int ni = 0; ni < 2; ++ni) {
#pragma unroll
      for (int v = 0; v < 8; ++v) {
        int grow = mBase + wm * 64 + mi * 16 + v + ((lane >> 4) << 3);
        int gcol = nBase + wn * 32 + ni * 16 + (lane & 15);
        float val = acc[mi][ni][v];
        if (bias) val += bias[gcol];
        if (base) val += base[(size_t)grow * N + gcol];
        if (cur2) val += cur2[grow * 2 + 0] * Weff[gcol]
                       + cur2[grow * 2 + 1] * Weff[N + gcol];
        C[(size_t)grow * N + gcol] = val;
      }
    }
  }
}

// ---- elementwise LSTM cell: gates g[BT,4H] (i,f,g,o), state c; emits h bf16 (+f32) ----
__global__ __launch_bounds__(256)
void lstm_cell(const float* __restrict__ g, float* __restrict__ c,
               bf16_t* __restrict__ hb, float* __restrict__ hf)
{
  int idx = blockIdx.x * blockDim.x + threadIdx.x;
  if (idx >= BT * HDIM) return;
  int row = idx >> 10;
  int j   = idx & (HDIM - 1);
  const float* gr = g + (size_t)row * H4;
  float ig = sigmoidf_(gr[j]);
  float fg = sigmoidf_(gr[HDIM + j]);
  float gg = tanhf(gr[2 * HDIM + j]);
  float og = sigmoidf_(gr[3 * HDIM + j]);
  float cn = fmaf(fg, c[idx], ig * gg);
  c[idx] = cn;
  float h = og * tanhf(cn);
  hb[idx] = f32_to_bf16(h);
  if (hf) hf[idx] = h;
}

// ---- output projection + decode: logits -> d_out[B,33,T,STEPS], cur[BT,2] ----
__global__ __launch_bounds__(64)
void outproj_argmax(const float* __restrict__ h1, const float* __restrict__ Wo,
                    const float* __restrict__ bo, float* __restrict__ out,
                    float* __restrict__ cur2, int step)
{
  __shared__ float hrow[HDIM];
  __shared__ float vals[NOUT];
  int row = blockIdx.x;
  int t = threadIdx.x;
#pragma unroll
  for (int i = 0; i < HDIM / 64; ++i)
    hrow[t + i * 64] = h1[(size_t)row * HDIM + t + i * 64];
  __syncthreads();
  if (t < NOUT) {
    float acc = bo[t];
#pragma unroll 4
    for (int k = 0; k < HDIM; ++k) acc = fmaf(hrow[k], Wo[k * NOUT + t], acc);
    vals[t] = acc;
    int b = row >> 6, tt = row & 63;
    out[(((size_t)b * NOUT + t) * 64 + tt) * NSTEP + step] = acc;
  }
  __syncthreads();
  if (t == 0) {
    float note = vals[0] > 0.0f ? 1.0f : 0.0f;     // sigmoid(x)>0.5 <=> x>0
    float best = vals[1]; int bidx = 0;
#pragma unroll 1
    for (int k = 2; k < NOUT; ++k) if (vals[k] > best) { best = vals[k]; bidx = k - 1; }
    cur2[row * 2 + 0] = note;
    cur2[row * 2 + 1] = (float)bidx / (float)(VB - 1);
  }
}

// ---- prep kernels ----
__global__ void transpose_cast_bf16(const float* __restrict__ W, bf16_t* __restrict__ Wt,
                                    int Kdim, int Ncols, int rowOff)
{
  int idx = blockIdx.x * blockDim.x + threadIdx.x;
  if (idx >= Kdim * Ncols) return;
  int k = idx % Kdim, n = idx / Kdim;
  Wt[idx] = f32_to_bf16(W[(size_t)(rowOff + k) * Ncols + n]);
}

__global__ void cast_to_bf16(const float* __restrict__ s, bf16_t* __restrict__ d, int n)
{
  int i = blockIdx.x * blockDim.x + threadIdx.x;
  if (i < n) d[i] = f32_to_bf16(s[i]);
}

__global__ void weff_bias0_kernel(const float* __restrict__ Wi, const float* __restrict__ bi,
                                  const float* __restrict__ Wih0,
                                  const float* __restrict__ bih0, const float* __restrict__ bhh0,
                                  float* __restrict__ Weff, float* __restrict__ bias0)
{
  int col = blockIdx.x * blockDim.x + threadIdx.x;
  if (col >= H4) return;
  float w0 = 0.f, w1 = 0.f, bv = 0.f;
  for (int k = 0; k < HDIM; ++k) {
    float w = Wih0[(size_t)k * H4 + col];
    w0 = fmaf(Wi[k], w, w0);
    w1 = fmaf(Wi[HDIM + k], w, w1);
    bv = fmaf(bi[k], w, bv);
  }
  Weff[col] = w0; Weff[H4 + col] = w1;
  bias0[col] = bv + bih0[col] + bhh0[col];
}

__global__ void add_vec(const float* a, const float* b, float* o, int n)
{
  int i = blockIdx.x * blockDim.x + threadIdx.x;
  if (i < n) o[i] = a[i] + b[i];
}

__global__ void init_state(float* c0, float* c1, bf16_t* h0b, bf16_t* h1b, float* h1f,
                           float* cur2, const float* __restrict__ sos)
{
  int idx = blockIdx.x * blockDim.x + threadIdx.x;
  if (idx < BT * HDIM) {
    c0[idx] = 0.f; c1[idx] = 0.f; h1f[idx] = 0.f;
    h0b[idx] = f32_to_bf16(0.f); h1b[idx] = f32_to_bf16(0.f);
  }
  if (idx < BT) { cur2[idx * 2 + 0] = sos[0]; cur2[idx * 2 + 1] = sos[1]; }
}

// -------------------------------------------------------------------------
extern "C" void kernel_launch(void* const* d_in, const int* in_sizes, int n_in,
                              void* d_out, int out_size, void* d_ws, size_t ws_size,
                              hipStream_t stream)
{
  (void)in_sizes; (void)n_in; (void)out_size; (void)ws_size;
  const float* x    = (const float*)d_in[0];
  const float* sos  = (const float*)d_in[1];
  const float* Wc   = (const float*)d_in[2];
  const float* bc   = (const float*)d_in[3];
  const float* Wi   = (const float*)d_in[4];
  const float* bi   = (const float*)d_in[5];
  const float* Wih0 = (const float*)d_in[6];
  const float* Whh0 = (const float*)d_in[7];
  const float* bih0 = (const float*)d_in[8];
  const float* bhh0 = (const float*)d_in[9];
  const float* Wih1 = (const float*)d_in[10];
  const float* Whh1 = (const float*)d_in[11];
  const float* bih1 = (const float*)d_in[12];
  const float* bhh1 = (const float*)d_in[13];
  const float* Wo   = (const float*)d_in[14];
  const float* bo   = (const float*)d_in[15];
  float* out = (float*)d_out;

  // workspace carve (256B aligned)
  char* p = (char*)d_ws;
  auto carve = [&](size_t bytes) -> char* {
    char* r = p; p += (bytes + 255) & ~(size_t)255; return r;
  };
  bf16_t* Whh0t   = (bf16_t*)carve((size_t)H4 * HDIM * 2);
  bf16_t* Wih1t   = (bf16_t*)carve((size_t)H4 * HDIM * 2);
  bf16_t* Whh1t   = (bf16_t*)carve((size_t)H4 * HDIM * 2);
  bf16_t* Wih0hiT = (bf16_t*)carve((size_t)H4 * HDIM * 2);
  bf16_t* WcT     = (bf16_t*)carve((size_t)HDIM * EDIM * 2);
  bf16_t* x_bf    = (bf16_t*)carve((size_t)BT * EDIM * 2);
  bf16_t* cond_bf = (bf16_t*)carve((size_t)BT * HDIM * 2);
  bf16_t* h0_bf   = (bf16_t*)carve((size_t)BT * HDIM * 2);
  bf16_t* h1_bf   = (bf16_t*)carve((size_t)BT * HDIM * 2);
  float* cond   = (float*)carve((size_t)BT * HDIM * 4);
  float* base0  = (float*)carve((size_t)BT * H4 * 4);
  float* g0     = (float*)carve((size_t)BT * H4 * 4);
  float* g1     = (float*)carve((size_t)BT * H4 * 4);
  float* c0     = (float*)carve((size_t)BT * HDIM * 4);
  float* c1     = (float*)carve((size_t)BT * HDIM * 4);
  float* h1f    = (float*)carve((size_t)BT * HDIM * 4);
  float* curb   = (float*)carve((size_t)BT * 2 * 4);
  float* Weff   = (float*)carve((size_t)2 * H4 * 4);
  float* bias0v = (float*)carve((size_t)H4 * 4);
  float* bias1v = (float*)carve((size_t)H4 * 4);

  const int TB = 256;
  // --- prep: weight transposes to bf16 [N,K] layout ---
  int nbig = (HDIM * H4 + TB - 1) / TB;
  transpose_cast_bf16<<<nbig, TB, 0, stream>>>(Whh0, Whh0t, HDIM, H4, 0);
  transpose_cast_bf16<<<nbig, TB, 0, stream>>>(Wih1, Wih1t, HDIM, H4, 0);
  transpose_cast_bf16<<<nbig, TB, 0, stream>>>(Whh1, Whh1t, HDIM, H4, 0);
  transpose_cast_bf16<<<nbig, TB, 0, stream>>>(Wih0, Wih0hiT, HDIM, H4, HDIM); // rows H..2H-1
  transpose_cast_bf16<<<(EDIM * HDIM + TB - 1) / TB, TB, 0, stream>>>(Wc, WcT, EDIM, HDIM, 0);
  cast_to_bf16<<<(BT * EDIM + TB - 1) / TB, TB, 0, stream>>>(x, x_bf, BT * EDIM);
  weff_bias0_kernel<<<(H4 + TB - 1) / TB, TB, 0, stream>>>(Wi, bi, Wih0, bih0, bhh0, Weff, bias0v);
  add_vec<<<(H4 + TB - 1) / TB, TB, 0, stream>>>(bih1, bhh1, bias1v, H4);
  init_state<<<(BT * HDIM + TB - 1) / TB, TB, 0, stream>>>(c0, c1, h0_bf, h1_bf, h1f, curb, sos);

  // cond = x @ Wc + bc        [512,512]x[512,1024]
  gemm_bf16_wmma<<<dim3(HDIM / 128, BT / 128), TB, 0, stream>>>(
      x_bf, WcT, nullptr, nullptr, bc, nullptr, nullptr, nullptr, cond, BT, HDIM, EDIM);
  cast_to_bf16<<<(BT * HDIM + TB - 1) / TB, TB, 0, stream>>>(cond, cond_bf, BT * HDIM);
  // base0 = cond @ Wih0[H:2H] + (bi@Wih0[:H] + bih0 + bhh0)
  gemm_bf16_wmma<<<dim3(H4 / 128, BT / 128), TB, 0, stream>>>(
      cond_bf, Wih0hiT, nullptr, nullptr, bias0v, nullptr, nullptr, nullptr, base0, BT, H4, HDIM);

  dim3 ggrid(H4 / 128, BT / 128);   // (32, 4)
  int  cgrid = (BT * HDIM + TB - 1) / TB;
  for (int step = 0; step < NSTEP; ++step) {
    // layer 0 gates: g0 = h0 @ Whh0 + base0 + cur @ Weff
    gemm_bf16_wmma<<<ggrid, TB, 0, stream>>>(
        h0_bf, Whh0t, nullptr, nullptr, nullptr, base0, curb, Weff, g0, BT, H4, HDIM);
    lstm_cell<<<cgrid, TB, 0, stream>>>(g0, c0, h0_bf, nullptr);
    // layer 1 gates: g1 = h0 @ Wih1 + h1 @ Whh1 + bias1
    gemm_bf16_wmma<<<ggrid, TB, 0, stream>>>(
        h0_bf, Wih1t, h1_bf, Whh1t, bias1v, nullptr, nullptr, nullptr, g1, BT, H4, HDIM);
    lstm_cell<<<cgrid, TB, 0, stream>>>(g1, c1, h1_bf, h1f);
    // logits + decode next cur
    outproj_argmax<<<BT, 64, 0, stream>>>(h1f, Wo, bo, out, curb, step);
  }
}
